// MultiSensor_TimeMIL_v1_46772193853794
// MI455X (gfx1250) — compile-verified, matrix operand order checked by reading, not run on hardware
//
#include <hip/hip_runtime.h>

// ---------------------------------------------------------------------------
// Types for CDNA5 WMMA
// ---------------------------------------------------------------------------
typedef _Float16 v16h __attribute__((ext_vector_type(16)));
typedef float    v8f  __attribute__((ext_vector_type(8)));
typedef float    v4f  __attribute__((ext_vector_type(4)));
typedef _Float16 v8h  __attribute__((ext_vector_type(8)));

#define GF_BIAS 1
#define GF_RELU 2
#define GF_RES  4

#define BN_INV_C 0.9999950000374997f
#define LSEQ 512

// ---------------------------------------------------------------------------
// Tensor Data Mover: 2D f32 tile -> LDS (clang-23 6-arg builtin:
// (uint32x4 g0, int32x8 g1, int32x4 g2, int32x4 g3, int32x8 g4, i32 cpol)).
// Group encodings per CDNA5 ISA 8.3/8.4. Row pitch in LDS = tile_d0 + 4 dwords
// via pad_interval=32dw / pad_amount=4dw (keeps rows 16B-aligned for b128).
// ---------------------------------------------------------------------------
#if defined(__has_builtin)
#if __has_builtin(__builtin_amdgcn_tensor_load_to_lds) && \
    __has_builtin(__builtin_amdgcn_s_wait_tensorcnt)
#define HAVE_TDM 1
#endif
#endif

#ifdef HAVE_TDM
typedef unsigned int tdm_v4u __attribute__((ext_vector_type(4)));
typedef int          tdm_v8i __attribute__((ext_vector_type(8)));
typedef int          tdm_v4i __attribute__((ext_vector_type(4)));

__device__ __forceinline__ void tdm_load_2d_f32(
    unsigned lds_byte_off, const float* gaddr,
    unsigned tensor_d0, unsigned tensor_d1, long long row_stride_elems,
    unsigned tile_d0, unsigned tile_d1)
{
    unsigned long long ga = (unsigned long long)(size_t)gaddr;
    tdm_v4u g0;
    g0[0] = 1u;                                    // count=1, user descriptor
    g0[1] = lds_byte_off;                          // lds_addr
    g0[2] = (unsigned)(ga & 0xffffffffu);          // global_addr lo
    g0[3] = (unsigned)((ga >> 32) & 0x01ffffffu)   // global_addr hi
          | 0x80000000u;                           // type=2 ("image")
    unsigned long long st = (unsigned long long)row_stride_elems;
    tdm_v8i g1;
    g1[0] = (int)((2u << 16)                       // data_size = 4B
                | (1u << 20)                       // pad_enable
                | (4u << 22)                       // pad_interval = 32 dwords
                | (3u << 25));                     // pad_amount = 4 dwords
    g1[1] = (int)((tensor_d0 & 0xffffu) << 16);    // tensor_dim0[15:0]
    g1[2] = (int)(((tensor_d0 >> 16) & 0xffffu) | ((tensor_d1 & 0xffffu) << 16));
    g1[3] = (int)(((tensor_d1 >> 16) & 0xffffu) | ((tile_d0 & 0xffffu) << 16));
    g1[4] = (int)(tile_d1 & 0xffffu);              // tile_dim1, tile_dim2=0
    g1[5] = (int)(unsigned)(st & 0xffffffffu);     // dim0 stride lo
    g1[6] = (int)(unsigned)((st >> 32) & 0xffffu); // dim0 stride hi
    g1[7] = 0;
    tdm_v4i gz4 = { 0, 0, 0, 0 };
    tdm_v8i gz8 = { 0, 0, 0, 0, 0, 0, 0, 0 };
    __builtin_amdgcn_tensor_load_to_lds(g0, g1, gz4, gz4, gz8, 0);
}
#endif

// ---------------------------------------------------------------------------
// Batched strided GEMM: C = act(alpha * A@B + bias + R)
// A(m,k) = A[off + zo*sAo + zi*sAi + m*aR + k*aC]   (f32)
// B(k,n) = B[off + zo*sBo + zi*sBi + k*bR + n*bC]
// C(m,n) = C[off + zo*sCo + zi*sCi + m*cR + n]
// Tile: 64(M) x 64(N) per 256-thread block; 8 waves, each 16(M) x 32(N).
// A staged f32 (TDM when aC==1), gathered with 4x b128 + packed cvt.
// B staged transposed f16 [n][k], gathered with 2x 16B loads per fragment.
// Lane fragment K-runs: {kb..kb+7} u {16+kb..16+kb+7}, kb = (lane>>4)*8.
// ---------------------------------------------------------------------------
#define AS_PITCH 36                    // f32 per A row (144B, 16B aligned)
#define AS_ELEMS (64 * AS_PITCH)       // 2304 f32
#define BS_PITCH 40                    // f16 per B row (80B, 16B aligned)

__global__ __launch_bounds__(256) void gemm_wmma(
    const float* __restrict__ A, const float* __restrict__ Bm,
    const float* __restrict__ bias, const float* __restrict__ Rm,
    float* __restrict__ C,
    int M, int N, int K,
    long long offA, long long sAo, long long sAi, int aR, int aC,
    long long offB, long long sBo, long long sBi, int bR, int bC,
    long long offC, long long sCo, long long sCi, int cR,
    long long offR, long long sRo, long long sRi, int rR,
    int binner, float alpha, int flags)
{
    // A region: 2304 f32; B region: 64*40 f16 = 2560 halves = 1280 f32 slots
    __shared__ __attribute__((aligned(16))) float lds_pool[AS_ELEMS + 1280];
    _Float16* BsH = (_Float16*)(lds_pool + AS_ELEMS);
#define AS_(m, k) lds_pool[(m) * AS_PITCH + (k)]
#define BT_(n, k) BsH[(n) * BS_PITCH + (k)]

    int z  = blockIdx.z;
    int zo = z / binner, zi = z % binner;
    const float* Ab = A  + offA + (long long)zo * sAo + (long long)zi * sAi;
    const float* Bb = Bm + offB + (long long)zo * sBo + (long long)zi * sBi;
    float*       Cb = C  + offC + (long long)zo * sCo + (long long)zi * sCi;
    const float* Rb = (flags & GF_RES)
        ? (Rm + offR + (long long)zo * sRo + (long long)zi * sRi) : (const float*)0;

    int tid = threadIdx.x, lane = tid & 31, wave = tid >> 5;
    int mBase = blockIdx.y * 64, nBase = blockIdx.x * 64;
    int mOff = (wave >> 1) * 16;   // 4 wave rows of 16
    int nOff = (wave & 1) * 32;    // 2 wave cols of 32

    bool tdm = false;
#ifdef HAVE_TDM
    tdm = (aC == 1);
#endif

    v8f acc0 = {}, acc1 = {};
    int kb = (lane >> 4) << 3;     // per-lane K run base
    int mr = mOff + (lane & 15);
    int c0 = nOff + (lane & 15);

    for (int k0 = 0; k0 < K; k0 += 32) {
#ifdef HAVE_TDM
        if (tdm) {
            if (wave == 0) {
                tdm_load_2d_f32(0u, Ab + (long long)mBase * aR + k0,
                                (unsigned)(K - k0), (unsigned)(M - mBase),
                                (long long)aR, 32u, 64u);
            }
        } else
#endif
        {
            if (mBase + 64 <= M && k0 + 32 <= K) {
                for (int idx = tid; idx < 64 * 32; idx += 256) {
                    int m = idx >> 5, k = idx & 31;
                    AS_(m, k) = Ab[(long long)(mBase + m) * aR + (long long)(k0 + k) * aC];
                }
            } else {
                for (int idx = tid; idx < 64 * 32; idx += 256) {
                    int m = idx >> 5, k = idx & 31;
                    int gm = mBase + m, gk = k0 + k;
                    AS_(m, k) = (gm < M && gk < K)
                        ? Ab[(long long)gm * aR + (long long)gk * aC] : 0.0f;
                }
            }
        }
        // B tile 32(K) x 64(N), stored transposed f16 [n][k]
        if (k0 + 32 <= K && nBase + 64 <= N) {
            for (int idx = tid; idx < 32 * 64; idx += 256) {
                int kk = idx >> 6, n = idx & 63;
                BT_(n, kk) = (_Float16)Bb[(long long)(k0 + kk) * bR + (long long)(nBase + n) * bC];
            }
        } else {
            for (int idx = tid; idx < 32 * 64; idx += 256) {
                int kk = idx >> 6, n = idx & 63;
                int gk = k0 + kk, gn = nBase + n;
                float v = (gk < K && gn < N)
                    ? Bb[(long long)gk * bR + (long long)gn * bC] : 0.0f;
                BT_(n, kk) = (_Float16)v;
            }
        }
        // speculative prefetch of next A K-slice (global_prefetch_b8)
        if (k0 + 32 < K) {
            const float* pf = Ab + (long long)(mBase + (tid >> 2)) * aR
                                 + (long long)(k0 + 32 + ((tid & 3) << 3)) * aC;
            __builtin_prefetch(pf, 0, 1);
        }
#ifdef HAVE_TDM
        if (tdm && wave == 0) __builtin_amdgcn_s_wait_tensorcnt(0);
#endif
        __syncthreads();

        // A fragment: 4x ds_load_b128 (f32) + packed cvt
        const float* arow = &lds_pool[mr * AS_PITCH];
        v4f a0 = *(const v4f*)(arow + kb);
        v4f a1 = *(const v4f*)(arow + kb + 4);
        v4f a2 = *(const v4f*)(arow + kb + 16);
        v4f a3 = *(const v4f*)(arow + kb + 20);
        v16h af;
        #pragma unroll
        for (int j = 0; j < 4; ++j) {
            af[j]      = (_Float16)a0[j];
            af[j + 4]  = (_Float16)a1[j];
            af[j + 8]  = (_Float16)a2[j];
            af[j + 12] = (_Float16)a3[j];
        }
        // B fragments: 2x 16B LDS loads each
        const _Float16* brow0 = &BT_(c0, 0);
        const _Float16* brow1 = &BT_(c0 + 16, 0);
        v8h b0lo = *(const v8h*)(brow0 + kb);
        v8h b0hi = *(const v8h*)(brow0 + 16 + kb);
        v8h b1lo = *(const v8h*)(brow1 + kb);
        v8h b1hi = *(const v8h*)(brow1 + 16 + kb);
        v16h bf0, bf1;
        #pragma unroll
        for (int j = 0; j < 8; ++j) {
            bf0[j]     = b0lo[j];
            bf0[j + 8] = b0hi[j];
            bf1[j]     = b1lo[j];
            bf1[j + 8] = b1hi[j];
        }
        acc0 = __builtin_amdgcn_wmma_f32_16x16x32_f16(
            false, af, false, bf0, (short)0, acc0, false, false);
        acc1 = __builtin_amdgcn_wmma_f32_16x16x32_f16(
            false, af, false, bf1, (short)0, acc1, false, false);
        __syncthreads();
    }

    int mHi = (lane >> 4) << 3;
    int nG0 = nBase + nOff + (lane & 15);
    int nG1 = nG0 + 16;
    #pragma unroll
    for (int r = 0; r < 8; ++r) {
        int mG = mBase + mOff + mHi + r;
        if (mG < M) {
            long long rowC = (long long)mG * cR;
            if (nG0 < N) {
                float v = acc0[r] * alpha;
                if (flags & GF_BIAS) v += bias[nG0];
                if (flags & GF_RES)  v += Rb[(long long)mG * rR + nG0];
                if (flags & GF_RELU) v = fmaxf(v, 0.0f);
                Cb[rowC + nG0] = v;
            }
            if (nG1 < N) {
                float v = acc1[r] * alpha;
                if (flags & GF_BIAS) v += bias[nG1];
                if (flags & GF_RES)  v += Rb[(long long)mG * rR + nG1];
                if (flags & GF_RELU) v = fmaxf(v, 0.0f);
                Cb[rowC + nG1] = v;
            }
        }
    }
#undef AS_
#undef BT_
}

// ---------------------------------------------------------------------------
// Conv1d (SAME) via per-tap WMMA over channel dim.
// X: [B, CI, 512] compact. W: [CO, CI, KS]. Y: [B, COT, 512] written at chOff.
// Block = 128 threads (4 waves): 32(CO) x 64(L); each wave 16(CO) x 32(L).
// Slab stored transposed [t][ci] f16 so x-fragments are 2x 16B LDS loads.
// ---------------------------------------------------------------------------
#define XS_PITCH 136   // f16 per slab row (272B, 16B aligned)

__global__ __launch_bounds__(128) void conv1d_wmma(
    const float* __restrict__ X, const float* __restrict__ Wt,
    float* __restrict__ Y,
    int CI, int KSz, int pad, int chOff, int COT)
{
    __shared__ __attribute__((aligned(16))) _Float16 xs[104][XS_PITCH];

    int b = blockIdx.z;
    int coBase = blockIdx.y * 32;
    int lBase  = blockIdx.x * 64;
    int CIp = (CI + 31) & ~31;
    int SW  = 64 + KSz - 1;
    int tid = threadIdx.x, lane = tid & 31, wave = tid >> 5;

    // fill transposed: read X coalesced along l, scatter into xs[t][ci]
    for (int idx = tid; idx < CIp * SW; idx += 128) {
        int ci = idx / SW, t = idx % SW;
        int l = lBase - pad + t;
        float v = (ci < CI && l >= 0 && l < LSEQ)
            ? X[((long long)b * CI + ci) * LSEQ + l] : 0.0f;
        xs[t][ci] = (_Float16)v;
    }
    __syncthreads();

    int coOff = (wave & 1) * 16;
    int lOff  = (wave >> 1) * 32;
    v8f acc0 = {}, acc1 = {};
    int kb = (lane >> 4) << 3;
    int mr = coBase + coOff + (lane & 15);  // output channel row for A frag
    int nr = lOff + (lane & 15);            // local time col for B frag

    for (int ci0 = 0; ci0 < CIp; ci0 += 32) {
        bool allIn = (ci0 + 32 <= CI);
        for (int t = 0; t < KSz; ++t) {
            v16h af, bf0, bf1;
            // weights: per-lane K-runs {kb..kb+7} u {16+kb..16+kb+7} over ci
            if (allIn) {
                const float* wrow = Wt + ((long long)mr * CI + ci0) * KSz + t;
                #pragma unroll
                for (int j = 0; j < 8; ++j) {
                    af[j]     = (_Float16)wrow[(long long)(kb + j) * KSz];
                    af[j + 8] = (_Float16)wrow[(long long)(16 + kb + j) * KSz];
                }
            } else {
                #pragma unroll
                for (int j = 0; j < 8; ++j) {
                    int ciA = ci0 + kb + j, ciB = ci0 + 16 + kb + j;
                    af[j]     = (_Float16)((ciA < CI)
                        ? Wt[((long long)mr * CI + ciA) * KSz + t] : 0.0f);
                    af[j + 8] = (_Float16)((ciB < CI)
                        ? Wt[((long long)mr * CI + ciB) * KSz + t] : 0.0f);
                }
            }
            // x fragments: contiguous channel runs in transposed slab
            const _Float16* x0 = &xs[nr + t][ci0 + kb];
            const _Float16* x1 = &xs[nr + 16 + t][ci0 + kb];
            v8h p0lo = *(const v8h*)(x0);
            v8h p0hi = *(const v8h*)(x0 + 16);
            v8h p1lo = *(const v8h*)(x1);
            v8h p1hi = *(const v8h*)(x1 + 16);
            #pragma unroll
            for (int j = 0; j < 8; ++j) {
                bf0[j]     = p0lo[j];
                bf0[j + 8] = p0hi[j];
                bf1[j]     = p1lo[j];
                bf1[j + 8] = p1hi[j];
            }
            acc0 = __builtin_amdgcn_wmma_f32_16x16x32_f16(
                false, af, false, bf0, (short)0, acc0, false, false);
            acc1 = __builtin_amdgcn_wmma_f32_16x16x32_f16(
                false, af, false, bf1, (short)0, acc1, false, false);
        }
    }

    int lG0 = lBase + lOff + (lane & 15);
    int lG1 = lG0 + 16;
    int mHi = (lane >> 4) << 3;
    #pragma unroll
    for (int r = 0; r < 8; ++r) {
        int co = coBase + coOff + mHi + r;
        long long rowY = ((long long)b * COT + chOff + co) * LSEQ;
        Y[rowY + lG0] = acc0[r];
        Y[rowY + lG1] = acc1[r];
    }
}

// ---------------------------------------------------------------------------
// Elementwise / reduction helpers
// ---------------------------------------------------------------------------
__global__ void sensor_transpose_k(float* __restrict__ out, const float* __restrict__ in,
                                   int S, int s, int V, size_t n) {
    size_t idx = (size_t)blockIdx.x * 256 + threadIdx.x;
    if (idx >= n) return;
    int l = (int)(idx & (LSEQ - 1));
    int c = (int)((idx >> 9) % V);
    int b = (int)(idx / ((size_t)V * LSEQ));
    out[idx] = in[(((size_t)b * S + s) * LSEQ + l) * V + c];
}

__global__ void maxpool3_k(float* __restrict__ out, const float* __restrict__ x, size_t n) {
    size_t idx = (size_t)blockIdx.x * 256 + threadIdx.x;
    if (idx >= n) return;
    int l = (int)(idx & (LSEQ - 1));
    float v = x[idx];
    if (l > 0)        v = fmaxf(v, x[idx - 1]);
    if (l < LSEQ - 1) v = fmaxf(v, x[idx + 1]);
    out[idx] = v;
}

__global__ void bn_relu_k(float* __restrict__ x, const float* __restrict__ g,
                          const float* __restrict__ b, size_t n) {
    size_t idx = (size_t)blockIdx.x * 256 + threadIdx.x;
    if (idx >= n) return;
    int c = (int)((idx >> 9) & 127);
    x[idx] = fmaxf(g[c] * (x[idx] * BN_INV_C) + b[c], 0.0f);
}

__global__ void add_bn_relu_k(float* __restrict__ x, const float* __restrict__ r,
                              const float* __restrict__ g, const float* __restrict__ b,
                              size_t n) {
    size_t idx = (size_t)blockIdx.x * 256 + threadIdx.x;
    if (idx >= n) return;
    int c = (int)((idx >> 9) & 127);
    x[idx] = fmaxf(x[idx] + g[c] * (r[idx] * BN_INV_C) + b[c], 0.0f);
}

__global__ void copy_channels_k(float* __restrict__ dst, const float* __restrict__ src,
                                int dstCT, int dOff, size_t n) {
    size_t idx = (size_t)blockIdx.x * 256 + threadIdx.x;
    if (idx >= n) return;
    int l = (int)(idx & (LSEQ - 1));
    int c = (int)((idx >> 9) & 127);
    int b = (int)(idx >> 16);
    dst[((size_t)b * dstCT + dOff + c) * LSEQ + l] = src[idx];
}

__global__ void copyn_k(float* __restrict__ dst, const float* __restrict__ src, size_t n) {
    size_t idx = (size_t)blockIdx.x * 256 + threadIdx.x;
    if (idx < n) dst[idx] = src[idx];
}

__global__ void bcast_cls_k(float* __restrict__ X, const float* __restrict__ cls) {
    int idx = blockIdx.x * 256 + threadIdx.x;
    if (idx >= 8 * 512) return;
    int b = idx >> 9, c = idx & 511;
    X[(size_t)b * 513 * 512 + c] = cls[c];
}

__global__ void copyrow0_k(float* __restrict__ dst, const float* __restrict__ src) {
    int idx = blockIdx.x * 256 + threadIdx.x;
    if (idx >= 8 * 512) return;
    int b = idx >> 9, c = idx & 511;
    dst[(size_t)b * 513 * 512 + c] = src[(size_t)b * 513 * 512 + c];
}

// Mexican-hat taps: TP[d][t] = sum over 3 wavelets
__global__ void taps_k(float* __restrict__ tp, const float* __restrict__ w1,
                       const float* __restrict__ w2, const float* __restrict__ w3) {
    int idx = blockIdx.x * 256 + threadIdx.x;
    if (idx >= 512 * 19) return;
    int d = idx / 19, t = idx % 19;
    const float Cw = 0.8673250705840776f;  // 2/(sqrt(3)*pi^0.25)
    const float* ws[3] = { w1, w2, w3 };
    float acc = 0.0f;
    for (int w = 0; w < 3; ++w) {
        float sc = ws[w][d], sh = ws[w][512 + d];
        float xs = (float)t - 9.0f - sh;
        float u = xs / sc;
        acc += Cw * (1.0f - u * u) * __expf(-0.5f * u * u) * rsqrtf(fabsf(sc));
    }
    tp[idx] = acc;
}

// Depthwise wavelet conv: POS[b][l][d] = sum_t TP[d][t] * X[b][1+l+t-9][d]
__global__ void dwwave_k(float* __restrict__ pos, const float* __restrict__ X,
                         const float* __restrict__ tp, size_t n) {
    size_t idx = (size_t)blockIdx.x * 256 + threadIdx.x;
    if (idx >= n) return;
    int d = (int)(idx & 511);
    int l = (int)((idx >> 9) & 511);
    int b = (int)(idx >> 18);
    float acc = 0.0f;
    for (int t = 0; t < 19; ++t) {
        int ll = l + t - 9;
        if (ll >= 0 && ll < 512)
            acc += tp[d * 19 + t] * X[((size_t)b * 513 + 1 + ll) * 512 + d];
    }
    pos[((size_t)b * 512 + l) * 512 + d] = acc;
}

__global__ __launch_bounds__(256) void ln_k(float* __restrict__ Y, const float* __restrict__ X,
                                            const float* __restrict__ g, const float* __restrict__ b) {
    int row = blockIdx.x;
    const float* p = X + (size_t)row * 512;
    float*       q = Y + (size_t)row * 512;
    __shared__ float red[256];
    int t = threadIdx.x;
    float a0 = p[t], a1 = p[t + 256];
    red[t] = a0 + a1; __syncthreads();
    for (int s = 128; s > 0; s >>= 1) { if (t < s) red[t] += red[t + s]; __syncthreads(); }
    float mu = red[0] * (1.0f / 512.0f); __syncthreads();
    float d0 = a0 - mu, d1 = a1 - mu;
    red[t] = d0 * d0 + d1 * d1; __syncthreads();
    for (int s = 128; s > 0; s >>= 1) { if (t < s) red[t] += red[t + s]; __syncthreads(); }
    float inv = rsqrtf(red[0] * (1.0f / 512.0f) + 1e-5f);
    q[t]       = g[t] * d0 * inv + b[t];
    q[t + 256] = g[t + 256] * d1 * inv + b[t + 256];
}

// pad: XP[b][r][c] = (r<255) ? 0 : XN[b][r-255][c]
__global__ void pad_k(float* __restrict__ XP, const float* __restrict__ XN, size_t n) {
    size_t idx = (size_t)blockIdx.x * 256 + threadIdx.x;
    if (idx >= n) return;
    int c = (int)(idx & 511);
    size_t rb = idx >> 9;
    int r = (int)(rb % 768);
    int b = (int)(rb / 768);
    XP[idx] = (r < 255) ? 0.0f : XN[((size_t)b * 513 + (r - 255)) * 512 + c];
}

__global__ __launch_bounds__(256) void softmax_k(float* __restrict__ X, int cols) {
    size_t row = blockIdx.x;
    float* p = X + row * (size_t)cols;
    __shared__ float red[256];
    int t = threadIdx.x;
    float m = -3.0e38f;
    for (int c = t; c < cols; c += 256) m = fmaxf(m, p[c]);
    red[t] = m; __syncthreads();
    for (int s = 128; s > 0; s >>= 1) { if (t < s) red[t] = fmaxf(red[t], red[t + s]); __syncthreads(); }
    m = red[0]; __syncthreads();
    float s = 0.0f;
    for (int c = t; c < cols; c += 256) { float e = __expf(p[c] - m); p[c] = e; s += e; }
    red[t] = s; __syncthreads();
    for (int st = 128; st > 0; st >>= 1) { if (t < st) red[t] += red[t + st]; __syncthreads(); }
    float inv = 1.0f / red[0];
    for (int c = t; c < cols; c += 256) p[c] *= inv;
}

// landmark means: out[b][h][i][d] = mean_{j<3} qkv[b][3i+j][off + h*64 + d]
__global__ void lmean_k(float* __restrict__ out, const float* __restrict__ qkv, int off) {
    size_t idx = (size_t)blockIdx.x * 256 + threadIdx.x;
    if (idx >= (size_t)8 * 8 * 256 * 64) return;
    int d = (int)(idx & 63);
    int i = (int)((idx >> 6) & 255);
    int h = (int)((idx >> 14) & 7);
    int b = (int)(idx >> 17);
    size_t base = ((size_t)b * 768) * 1536 + off + h * 64 + d;
    float s = qkv[base + (size_t)(3 * i) * 1536]
            + qkv[base + (size_t)(3 * i + 1) * 1536]
            + qkv[base + (size_t)(3 * i + 2) * 1536];
    out[idx] = s * (1.0f / 3.0f);
}

__global__ void zero2_k(float* sc) { if (threadIdx.x < 2 && blockIdx.x == 0) sc[threadIdx.x] = 0.0f; }

__global__ __launch_bounds__(256) void rowsummax_k(const float* __restrict__ A2, float* __restrict__ sc) {
    size_t row = blockIdx.x;           // 64*256 rows
    const float* p = A2 + row * 256;
    __shared__ float red[256];
    int t = threadIdx.x;
    red[t] = fabsf(p[t]); __syncthreads();
    for (int s = 128; s > 0; s >>= 1) { if (t < s) red[t] += red[t + s]; __syncthreads(); }
    if (t == 0) atomicMax((int*)sc, __float_as_int(red[0]));
}

__global__ __launch_bounds__(256) void colsummax_k(const float* __restrict__ A2, float* __restrict__ sc) {
    int bh  = blockIdx.x >> 8;
    int col = blockIdx.x & 255;
    __shared__ float red[256];
    int t = threadIdx.x;
    red[t] = fabsf(A2[((size_t)bh << 16) + ((size_t)t << 8) + col]);
    __syncthreads();
    for (int s = 128; s > 0; s >>= 1) { if (t < s) red[t] += red[t + s]; __syncthreads(); }
    if (t == 0) atomicMax((int*)sc, __float_as_int(red[0]));
}

// Z = A2^T / (sc0*sc1)   per (b,h)
__global__ void tscale_k(float* __restrict__ Z, const float* __restrict__ A2,
                         const float* __restrict__ sc, size_t n) {
    size_t idx = (size_t)blockIdx.x * 256 + threadIdx.x;
    if (idx >= n) return;
    size_t bh = idx >> 16;
    int r = (int)(idx & 65535);
    int i = r >> 8, j = r & 255;
    float inv = 1.0f / (sc[0] * sc[1]);
    Z[idx] = A2[(bh << 16) + ((size_t)j << 8) + i] * inv;
}

// out = a*I - in  (elementwise, alias-safe)
__global__ void negdiag_k(float* __restrict__ out, const float* __restrict__ in,
                          float a, size_t n) {
    size_t idx = (size_t)blockIdx.x * 256 + threadIdx.x;
    if (idx >= n) return;
    int r = (int)(idx & 65535);
    int i = r >> 8, j = r & 255;
    out[idx] = ((i == j) ? a : 0.0f) - in[idx];
}

// OC[b][n][h*64+dh] += sum_t rw[h][t] * v[b][n+t-16][h*64+dh]
__global__ void attres_k(float* __restrict__ OC, const float* __restrict__ qkv,
                         const float* __restrict__ rw, size_t n) {
    size_t idx = (size_t)blockIdx.x * 256 + threadIdx.x;
    if (idx >= n) return;
    int c = (int)(idx & 511);
    size_t rb = idx >> 9;
    int r = (int)(rb % 768);
    int b = (int)(rb / 768);
    int h = c >> 6;
    float acc = 0.0f;
    for (int t = 0; t < 33; ++t) {
        int nn = r + t - 16;
        if (nn >= 0 && nn < 768)
            acc += rw[h * 33 + t] * qkv[((size_t)b * 768 + nn) * 1536 + 1024 + c];
    }
    OC[idx] += acc;
}

// ---------------------------------------------------------------------------
// Host orchestration
// ---------------------------------------------------------------------------
extern "C" void kernel_launch(void* const* d_in, const int* in_sizes, int n_in,
                              void* d_out, int out_size, void* d_ws, size_t ws_size,
                              hipStream_t stream) {
    (void)in_sizes; (void)n_in; (void)out_size; (void)ws_size;

    // --- input leaf mapping (top-level insertion order; params pytree sorted) ---
    enum {
        IN_IMU = 0, IN_THM = 1, IN_TOF = 2,
        P_CLS = 3, P_FC1W = 4, P_FC1B = 5, P_FC2W = 6, P_FC2B = 7,
        P_IMU = 8,            // 47 leaves
        P_L1 = 55, P_L2 = 61, // per-layer: Wo,+1 Wqkv,+2 bo,+3 ln_b,+4 ln_g,+5 res_w
        P_PROJW = 67, P_PROJB = 68,
        P_THM = 69,           // 46 leaves
        P_TOF = 115,          // 47 leaves
        P_WAVE1 = 162, P_WAVE1_ = 163, P_WAVE2 = 164, P_WAVE2_ = 165,
        P_WAVE3 = 166, P_WAVE3_ = 167,
        P_WE1W = 168, P_WE1B = 169, P_WE2W = 170, P_WE2B = 171
    };
    auto F = [&](int i) { return (const float*)d_in[i]; };

    // --- workspace carve-up (floats) ---
    float* ws = (float*)d_ws;
    size_t off = 0;
    auto alloc = [&](size_t nel) { float* p = ws + off; off += (nel + 63) & ~(size_t)63; return p; };
    const size_t SB = (size_t)8 * 128 * 512;
    float *XA = alloc(SB), *XB = alloc(SB), *P0 = alloc(SB), *P1 = alloc(SB);
    float *XM = alloc(SB), *XR = alloc(SB), *XRES = alloc(SB);
    float *CONC = alloc((size_t)8 * 1408 * 512);
    float *X0 = alloc((size_t)8 * 513 * 512), *X1 = alloc((size_t)8 * 513 * 512);
    float *XN = alloc((size_t)8 * 513 * 512);
    float *XP = alloc((size_t)8 * 768 * 512);
    float *QKV = alloc((size_t)8 * 768 * 1536);
    float *QL = alloc((size_t)64 * 256 * 64), *KL = alloc((size_t)64 * 256 * 64);
    float *A1 = alloc((size_t)64 * 768 * 256);
    float *A2 = alloc((size_t)64 * 256 * 256);
    float *A3 = alloc((size_t)64 * 256 * 768);
    float *Zb = alloc((size_t)64 * 256 * 256), *Z2 = alloc((size_t)64 * 256 * 256);
    float *XZ = alloc((size_t)64 * 256 * 256), *WB = alloc((size_t)64 * 256 * 256);
    float *W2 = alloc((size_t)64 * 256 * 256);
    float *A3V = alloc((size_t)64 * 256 * 64), *T2 = alloc((size_t)64 * 256 * 64);
    float *OC = alloc((size_t)8 * 768 * 512);
    float *POS = alloc((size_t)8 * 512 * 512);
    float *TP = alloc(512 * 19);
    float *SC = alloc(64);
    float *HB = alloc(8 * 512);

    auto blks = [](size_t nel) { return dim3((unsigned)((nel + 255) / 256)); };

    auto gemm = [&](const float* A, const float* B, const float* bias, const float* R, float* C,
                    int M, int N, int K,
                    long long oA, long long sAo, long long sAi, int aR, int aC,
                    long long oB, long long sBo, long long sBi, int bR, int bC,
                    long long oC, long long sCo, long long sCi, int cR,
                    long long oR, long long sRo, long long sRi, int rR,
                    int bo, int bi, float alpha, int flags) {
        dim3 g((N + 63) / 64, (M + 63) / 64, bo * bi);
        gemm_wmma<<<g, 256, 0, stream>>>(A, B, bias, R, C, M, N, K,
                                         oA, sAo, sAi, aR, aC,
                                         oB, sBo, sBi, bR, bC,
                                         oC, sCo, sCi, cR,
                                         oR, sRo, sRi, rR,
                                         bi, alpha, flags);
    };

    auto conv = [&](const float* X, const float* W, float* Y,
                    int CI, int CO, int KSz, int pad, int chOff, int COT) {
        dim3 g(8, CO / 32, 8);
        conv1d_wmma<<<g, 128, 0, stream>>>(X, W, Y, CI, KSz, pad, chOff, COT);
    };

    // ---------------- inception extractors ----------------
    auto inception = [&](const float* raw, int S, int s, int CI0, int fbase, bool b0, int concOff) {
        size_t n0 = (size_t)8 * CI0 * 512;
        sensor_transpose_k<<<blks(n0), 256, 0, stream>>>(XA, raw, S, s, CI0, n0);
        float* x = XA; int xCI = CI0;
        float* res = XA; int resCI = CI0;
        int m0 = b0 ? 7 : 6;
        for (int d = 0; d < 6; ++d) {
            int mb = fbase + (d == 0 ? 0 : m0 + 7 * (d - 1));
            bool hasB = (d > 0) || b0;
            int cW = mb + (hasB ? 3 : 2);
            float* cat = (d & 1) ? P1 : P0;
            const float* xb; int cb;
            if (hasB) { conv(x, F(mb + 2), XB, xCI, 32, 1, 0, 0, 32); xb = XB; cb = 32; }
            else      { xb = x; cb = xCI; }
            conv(xb, F(cW + 0), cat, cb, 32, 39, 19, 0, 128);
            conv(xb, F(cW + 1), cat, cb, 32, 19, 9, 32, 128);
            conv(xb, F(cW + 2), cat, cb, 32, 9, 4, 64, 128);
            size_t nmp = (size_t)8 * xCI * 512;
            maxpool3_k<<<blks(nmp), 256, 0, stream>>>(XM, x, nmp);
            conv(XM, F(cW + 3), cat, xCI, 32, 1, 0, 96, 128);
            bn_relu_k<<<blks(SB), 256, 0, stream>>>(cat, F(mb + 1), F(mb + 0), SB);
            x = cat; xCI = 128;
            if (d % 3 == 2) {
                int sb = fbase + m0 + 35 + (d == 2 ? 0 : 3);
                const float* r;
                if (d == 2) { conv(res, F(sb + 2), XR, resCI, 128, 1, 0, 0, 128); r = XR; }
                else        { r = res; }
                add_bn_relu_k<<<blks(SB), 256, 0, stream>>>(x, r, F(sb + 1), F(sb + 0), SB);
                copyn_k<<<blks(SB), 256, 0, stream>>>(XRES, x, SB);
                res = XRES; resCI = 128;
            }
        }
        copy_channels_k<<<blks(SB), 256, 0, stream>>>(CONC, x, 1408, concOff, SB);
    };

    inception(F(IN_IMU), 1, 0, 7, P_IMU, true, 0);
    for (int s = 0; s < 5; ++s) inception(F(IN_TOF), 5, s, 64, P_TOF, true, 128 * (1 + s));
    for (int s = 0; s < 5; ++s) inception(F(IN_THM), 5, s, 1, P_THM, false, 128 * (6 + s));

    // ---------------- projection + cls token ----------------
    bcast_cls_k<<<blks(8 * 512), 256, 0, stream>>>(X0, F(P_CLS));
    gemm(CONC, F(P_PROJW), F(P_PROJB), 0, X0,
         512, 512, 1408,
         0, (long long)1408 * 512, 0, 1, 512,
         0, 0, 0, 1, 1408,
         512, (long long)513 * 512, 0, 512,
         0, 0, 0, 0,
         8, 1, 1.0f, GF_BIAS);

    // ---------------- wavelet encoding ----------------
    auto wavelet = [&](float* Xin, float* Xout, int w1, int w2, int w3, int Wi, int bi) {
        taps_k<<<blks(512 * 19), 256, 0, stream>>>(TP, F(w1), F(w2), F(w3));
        size_t np = (size_t)8 * 512 * 512;
        dwwave_k<<<blks(np), 256, 0, stream>>>(POS, Xin, TP, np);
        copyrow0_k<<<blks(8 * 512), 256, 0, stream>>>(Xout, Xin);
        gemm(POS, F(Wi), F(bi), Xin, Xout,
             512, 512, 512,
             0, (long long)512 * 512, 0, 512, 1,
             0, 0, 0, 1, 512,
             512, (long long)513 * 512, 0, 512,
             512, (long long)513 * 512, 0, 512,
             8, 1, 1.0f, GF_BIAS | GF_RES);
    };

    // ---------------- nystrom transformer layer ----------------
    auto translayer = [&](float* Xin, float* Xout, int lb) {
        ln_k<<<8 * 513, 256, 0, stream>>>(XN, Xin, F(lb + 4), F(lb + 3));
        size_t npad = (size_t)8 * 768 * 512;
        pad_k<<<blks(npad), 256, 0, stream>>>(XP, XN, npad);
        gemm(XP, F(lb + 1), 0, 0, QKV, 768, 1536, 512,
             0, (long long)768 * 512, 0, 512, 1,
             0, 0, 0, 1, 512,
             0, (long long)768 * 1536, 0, 1536,
             0, 0, 0, 0, 8, 1, 1.0f, 0);
        size_t nlm = (size_t)64 * 256 * 64;
        lmean_k<<<blks(nlm), 256, 0, stream>>>(QL, QKV, 0);
        lmean_k<<<blks(nlm), 256, 0, stream>>>(KL, QKV, 512);
        // a1 = softmax(q @ kl^T * s)
        gemm(QKV, KL, 0, 0, A1, 768, 256, 64,
             0, (long long)768 * 1536, 64, 1536, 1,
             0, (long long)8 * 256 * 64, (long long)256 * 64, 1, 64,
             0, (long long)8 * 768 * 256, (long long)768 * 256, 256,
             0, 0, 0, 0, 8, 8, 0.125f, 0);
        softmax_k<<<64 * 768, 256, 0, stream>>>(A1, 256);
        // a2 = softmax(ql @ kl^T * s)
        gemm(QL, KL, 0, 0, A2, 256, 256, 64,
             0, (long long)8 * 256 * 64, (long long)256 * 64, 64, 1,
             0, (long long)8 * 256 * 64, (long long)256 * 64, 1, 64,
             0, (long long)8 * 256 * 256, (long long)256 * 256, 256,
             0, 0, 0, 0, 8, 8, 0.125f, 0);
        softmax_k<<<64 * 256, 256, 0, stream>>>(A2, 256);
        // a3 = softmax(ql @ k^T * s)
        gemm(QL, QKV, 0, 0, A3, 256, 768, 64,
             0, (long long)8 * 256 * 64, (long long)256 * 64, 64, 1,
             512, (long long)768 * 1536, 64, 1, 1536,
             0, (long long)8 * 256 * 768, (long long)256 * 768, 768,
             0, 0, 0, 0, 8, 8, 0.125f, 0);
        softmax_k<<<64 * 256, 256, 0, stream>>>(A3, 768);
        // pinv(a2) via Newton-Schulz
        zero2_k<<<1, 64, 0, stream>>>(SC);
        rowsummax_k<<<64 * 256, 256, 0, stream>>>(A2, SC);
        colsummax_k<<<64 * 256, 256, 0, stream>>>(A2, SC + 1);
        size_t nz = (size_t)64 * 256 * 256;
        tscale_k<<<blks(nz), 256, 0, stream>>>(Zb, A2, SC, nz);
        float *z = Zb, *z2 = Z2;
        for (int it = 0; it < 6; ++it) {
            gemm(A2, z, 0, 0, XZ, 256, 256, 256,
                 0, 65536, 0, 256, 1, 0, 65536, 0, 256, 1,
                 0, 65536, 0, 256, 0, 0, 0, 0, 64, 1, 1.0f, 0);
            negdiag_k<<<blks(nz), 256, 0, stream>>>(WB, XZ, 7.0f, nz);
            gemm(XZ, WB, 0, 0, W2, 256, 256, 256,
                 0, 65536, 0, 256, 1, 0, 65536, 0, 256, 1,
                 0, 65536, 0, 256, 0, 0, 0, 0, 64, 1, 1.0f, 0);
            negdiag_k<<<blks(nz), 256, 0, stream>>>(W2, W2, 15.0f, nz);
            gemm(XZ, W2, 0, 0, WB, 256, 256, 256,
                 0, 65536, 0, 256, 1, 0, 65536, 0, 256, 1,
                 0, 65536, 0, 256, 0, 0, 0, 0, 64, 1, 1.0f, 0);
            negdiag_k<<<blks(nz), 256, 0, stream>>>(WB, WB, 13.0f, nz);
            gemm(z, WB, 0, 0, z2, 256, 256, 256,
                 0, 65536, 0, 256, 1, 0, 65536, 0, 256, 1,
                 0, 65536, 0, 256, 0, 0, 0, 0, 64, 1, 0.25f, 0);
            float* tmp = z; z = z2; z2 = tmp;
        }
        // a3v = a3 @ v
        gemm(A3, QKV, 0, 0, A3V, 256, 64, 768,
             0, (long long)8 * 256 * 768, (long long)256 * 768, 768, 1,
             1024, (long long)768 * 1536, 64, 1536, 1,
             0, (long long)8 * 256 * 64, (long long)256 * 64, 64,
             0, 0, 0, 0, 8, 8, 1.0f, 0);
        // t2 = pinv(a2) @ a3v
        gemm(z, A3V, 0, 0, T2, 256, 64, 256,
             0, (long long)8 * 65536, 65536, 256, 1,
             0, (long long)8 * 256 * 64, (long long)256 * 64, 64, 1,
             0, (long long)8 * 256 * 64, (long long)256 * 64, 64,
             0, 0, 0, 0, 8, 8, 1.0f, 0);
        // OC[b][n][h*64+dh] = a1 @ t2  (heads concatenated via cR=512 strided write)
        gemm(A1, T2, 0, 0, OC, 768, 64, 256,
             0, (long long)8 * 768 * 256, (long long)768 * 256, 256, 1,
             0, (long long)8 * 256 * 64, (long long)256 * 64, 64, 1,
             0, (long long)768 * 512, 64, 512,
             0, 0, 0, 0, 8, 8, 1.0f, 0);
        // depthwise residual conv on v, added into OC
        attres_k<<<blks(npad), 256, 0, stream>>>(OC, QKV, F(lb + 5), npad);
        // out proj on last 513 rows + residual
        gemm(OC, F(lb), F(lb + 2), Xin, Xout, 513, 512, 512,
             (long long)255 * 512, (long long)768 * 512, 0, 512, 1,
             0, 0, 0, 1, 512,
             0, (long long)513 * 512, 0, 512,
             0, (long long)513 * 512, 0, 512,
             8, 1, 1.0f, GF_BIAS | GF_RES);
    };

    wavelet(X0, X1, P_WAVE1, P_WAVE2, P_WAVE3, P_WE1W, P_WE1B);
    translayer(X1, X0, P_L1);
    wavelet(X0, X1, P_WAVE1_, P_WAVE2_, P_WAVE3_, P_WE2W, P_WE2B);
    translayer(X1, X0, P_L2);

    // ---------------- classifier head ----------------
    gemm(X0, F(P_FC1W), F(P_FC1B), 0, HB, 8, 512, 512,
         0, 0, 0, 513 * 512, 1,
         0, 0, 0, 1, 512,
         0, 0, 0, 512,
         0, 0, 0, 0, 1, 1, 1.0f, GF_BIAS | GF_RELU);
    gemm(HB, F(P_FC2W), F(P_FC2B), 0, (float*)d_out, 8, 18, 512,
         0, 0, 0, 512, 1,
         0, 0, 0, 1, 512,
         0, 0, 0, 18,
         0, 0, 0, 0, 1, 1, 1.0f, GF_BIAS);
}
